// MultiheadAttention_ViTNO_36490042147155
// MI455X (gfx1250) — matrix operand order repaired and checked by reading, not compiled, not run on hardware
//
#include <hip/hip_runtime.h>

typedef __attribute__((ext_vector_type(16))) __bf16 v16bf;
typedef __attribute__((ext_vector_type(8)))  float  v8f;
typedef __attribute__((ext_vector_type(4)))  unsigned int v4u;
typedef __attribute__((ext_vector_type(8)))  int  v8i;
typedef __attribute__((ext_vector_type(4)))  int  v4i;

union BF16Frag { v16bf v; __bf16 h[16]; };

// order-preserving float -> uint key (for deterministic LDS u32 max atomics)
__device__ __forceinline__ unsigned f32_key(float f) {
  union { float f; unsigned u; } a; a.f = f;
  return (a.u & 0x80000000u) ? ~a.u : (a.u | 0x80000000u);
}
__device__ __forceinline__ float key_f32(unsigned k) {
  unsigned u = (k & 0x80000000u) ? (k ^ 0x80000000u) : ~k;
  union { unsigned u; float f; } a; a.u = u; return a.f;
}
__device__ __forceinline__ v8f wmma_bf16(v16bf a, v16bf b, v8f c) {
  // (neg_a, A, neg_b, B, c_mod, C, reuse_a, reuse_b)
  return __builtin_amdgcn_wmma_f32_16x16x32_bf16(false, a, false, b, (short)0, c,
                                                 false, false);
}

// LDS layout for the attention kernel (dynamic shared, no static LDS):
//   [0,       131072) : attn  (256x256 bf16)
//   [131072,  132096) : rowmaxK (256 u32)
//   [132096,  133120) : rowsum  (256 f32)
//   [133120,  264192) : V stage (256 rows x 256 cols bf16), TDM destination
#define ATTN_OFF    0u
#define ROWMAX_OFF  131072u
#define ROWSUM_OFF  132096u
#define VSTAGE_OFF  133120u
#define LDS_BYTES   264192u

// Issue a TDM 2D tile load: rows x cols bf16 from global (row stride in
// elements) into LDS at lds_off. Built per CDNA5 ISA ch.8 D# layout.
__device__ __forceinline__ void tdm_load_tile_bf16(
    const void* gptr, unsigned lds_off, unsigned rows, unsigned cols,
    unsigned row_stride_elems, unsigned tensor_cols) {
  unsigned long long ga = (unsigned long long)gptr;
  v4u g0;
  g0.x = 1u;                                         // count=1 (valid D#)
  g0.y = lds_off;                                    // lds_addr
  g0.z = (unsigned)(ga & 0xFFFFFFFFu);               // global_addr[31:0]
  g0.w = (unsigned)((ga >> 32) & 0x01FFFFFFu)        // global_addr[56:32]
         | (2u << 30);                               // type=2 ("image")
  v8i g1;
  g1[0] = (int)(1u << 16);                           // data_size=1 -> 2 bytes
  g1[1] = (int)((tensor_cols & 0xFFFFu) << 16);      // tensor_dim0[15:0]
  g1[2] = (int)(((tensor_cols >> 16) & 0xFFFFu)      // tensor_dim0[31:16]
                | ((rows & 0xFFFFu) << 16));         // tensor_dim1[15:0]
  g1[3] = (int)((cols & 0xFFFFu) << 16);             // tile_dim0
  g1[4] = (int)(rows & 0xFFFFu);                     // tile_dim1 (tile_dim2=0)
  g1[5] = (int)row_stride_elems;                     // tensor_dim0_stride[31:0]
  g1[6] = 0;                                         // stride hi / dim1_stride
  g1[7] = 0;
  v4i z4 = {0, 0, 0, 0};
#if __clang_major__ >= 23
  v8i z8 = {0, 0, 0, 0, 0, 0, 0, 0};
  __builtin_amdgcn_tensor_load_to_lds(g0, g1, z4, z4, z8, 0);
#else
  __builtin_amdgcn_tensor_load_to_lds(g0, g1, z4, z4, 0);
#endif
}

// ---------------------------------------------------------------------------
// Kernel 0: one-time f32 -> bf16 conversion of the four weight matrices.
// ---------------------------------------------------------------------------
__global__ __launch_bounds__(256) void k_cvt_weights(
    const float* __restrict__ Wq, const float* __restrict__ Wk,
    const float* __restrict__ Wv, const float* __restrict__ Wo,
    __bf16* __restrict__ out) {
  const int i = blockIdx.x * 256 + threadIdx.x;  // 0..65535
  out[i]           = (__bf16)Wq[i];
  out[i + 65536]   = (__bf16)Wk[i];
  out[i + 131072]  = (__bf16)Wv[i];
  out[i + 196608]  = (__bf16)Wo[i];
}

// ---------------------------------------------------------------------------
// Kernel 1: fused QKV projection. x:[M=131072,256] f32, W (bf16) from ws.
// Writes Qh/Kh/Vh bf16 in head-major layout [b][n][p][c], c=(x*8+y)*32+dk,
// model channel j -> head n=j%8, dk=j/8.
// ---------------------------------------------------------------------------
__global__ __launch_bounds__(256) void k_qkv_proj(
    const float* __restrict__ x,
    const __bf16* __restrict__ WqB, const float* __restrict__ bq,
    const __bf16* __restrict__ WkB, const float* __restrict__ bk,
    const __bf16* __restrict__ WvB, const float* __restrict__ bv,
    __bf16* __restrict__ Qh, __bf16* __restrict__ Kh,
    __bf16* __restrict__ Vh)
{
  const int wave = threadIdx.x >> 5;
  const int lane = threadIdx.x & 31;
  const int lr = lane & 15, hi = lane >> 4;
  const int m0 = blockIdx.x * 128 + wave * 16;

  // Preload A fragments for the whole K=256 (8 chunks of 32).
  // ISA 16-bit A 16x32 layout: lane<16 holds M=lane, K={0..7}+{16..23};
  // lane>=16 holds same M with K offset +8.
  BF16Frag A[8];
  const float* xrow = x + (size_t)(m0 + lr) * 256;
  __builtin_prefetch(xrow, 0, 1);
#pragma unroll
  for (int kk = 0; kk < 8; ++kk) {
    const int kb = kk * 32 + hi * 8;
#pragma unroll
    for (int i = 0; i < 8; ++i) {
      A[kk].h[i]     = (__bf16)xrow[kb + i];
      A[kk].h[8 + i] = (__bf16)xrow[kb + 16 + i];
    }
  }

  for (int nt = 0; nt < 16; ++nt) {
    const int j = nt * 16 + lr;                        // output column (lane=N)
    const __bf16* wq = WqB + (size_t)j * 256 + hi * 8; // B[k][n] = W[j][k]
    const __bf16* wk = WkB + (size_t)j * 256 + hi * 8;
    const __bf16* wv = WvB + (size_t)j * 256 + hi * 8;
    v8f cq = {}, ck = {}, cv = {};
#pragma unroll
    for (int kk = 0; kk < 8; ++kk) {
      BF16Frag Bq, Bk, Bv;
#pragma unroll
      for (int i = 0; i < 16; ++i) {
        Bq.h[i] = wq[kk * 32 + i + (i & 8)];
        Bk.h[i] = wk[kk * 32 + i + (i & 8)];
        Bv.h[i] = wv[kk * 32 + i + (i & 8)];
      }
      cq = wmma_bf16(A[kk].v, Bq.v, cq);
      ck = wmma_bf16(A[kk].v, Bk.v, ck);
      cv = wmma_bf16(A[kk].v, Bv.v, cv);
    }
    const float bqv = bq[j], bkv = bk[j], bvv = bv[j];
    const int n = j & 7, dk = j >> 3;
#pragma unroll
    for (int v = 0; v < 8; ++v) {
      const size_t m = (size_t)(m0 + v + 8 * hi);  // D: M = v + 8*lane[4]
      const int bb  = (int)(m >> 14);              // / (256*64)
      const int rem = (int)(m & 16383);
      const int p = rem >> 6, xy = rem & 63;
      const size_t hidx =
          (((size_t)bb * 8 + n) * 256 + p) * 2048 + (size_t)xy * 32 + dk;
      Qh[hidx] = (__bf16)(cq[v] + bqv);
      Kh[hidx] = (__bf16)(ck[v] + bkv);
      Vh[hidx] = (__bf16)(cv[v] + bvv);
    }
  }
}

// ---------------------------------------------------------------------------
// Kernel 2: attention per (b, head). scores = Q.K^T/64, softmax over q,
// O = attn.V. attn kept as bf16 in LDS; V staged into LDS 256-col slabs by
// the Tensor Data Mover; row max via ds u32-max atomics (order-independent);
// row sums deterministic. Output merged bf16 [b][p][xy][d], d = dk*8+head.
// ---------------------------------------------------------------------------
__global__ __launch_bounds__(256) void k_attention(
    const __bf16* __restrict__ Qh,
    const __bf16* __restrict__ Kh,
    const __bf16* __restrict__ Vh,
    __bf16* __restrict__ Am)
{
  extern __shared__ char smem[];
  __bf16* attn      = (__bf16*)(smem + ATTN_OFF);
  unsigned* rowmaxK = (unsigned*)(smem + ROWMAX_OFF);
  float* rowsum     = (float*)(smem + ROWSUM_OFF);
  const __bf16* vst = (const __bf16*)(smem + VSTAGE_OFF);

  const int tid = threadIdx.x;
  const int wave = tid >> 5, lane = tid & 31;
  const int lr = lane & 15, hi = lane >> 4;
  const int bb = blockIdx.x >> 3, head = blockIdx.x & 7;
  const size_t hoff = (size_t)blockIdx.x * 256 * 2048;
  const __bf16* Q = Qh + hoff;
  const __bf16* K = Kh + hoff;
  const __bf16* V = Vh + hoff;

  rowmaxK[tid] = 0u;        // key 0 < key(-inf): identity for u32 max
  rowsum[tid]  = 0.0f;
  __syncthreads();

  const float inv_scale = 1.0f / 64.0f;

  // Pass 1: S = Q K^T / 64; track row max (u32-key ds atomics).
  for (int t = wave; t < 256; t += 8) {
    const int p0 = (t >> 4) * 16, q0 = (t & 15) * 16;
    const __bf16* qrow = Q + (size_t)(p0 + lr) * 2048;
    const __bf16* krow = K + (size_t)(q0 + lr) * 2048;
    __builtin_prefetch(krow, 0, 0);
    v8f acc = {};
    for (int c0 = 0; c0 < 2048; c0 += 32) {
      BF16Frag a, b;
#pragma unroll
      for (int i = 0; i < 8; ++i) {
        a.h[i]     = qrow[c0 + hi * 8 + i];
        a.h[8 + i] = qrow[c0 + 16 + hi * 8 + i];
      }
#pragma unroll
      for (int i = 0; i < 16; ++i) b.h[i] = krow[c0 + hi * 8 + i + (i & 8)];
      acc = wmma_bf16(a.v, b.v, acc);
    }
#pragma unroll
    for (int v = 0; v < 8; ++v) {
      const int p = p0 + v + 8 * hi;
      atomicMax(&rowmaxK[p], f32_key(acc[v] * inv_scale));
    }
  }
  __syncthreads();

  // Pass 2: recompute S, exponentiate, store bf16 attn to LDS.
  for (int t = wave; t < 256; t += 8) {
    const int p0 = (t >> 4) * 16, q0 = (t & 15) * 16;
    const __bf16* qrow = Q + (size_t)(p0 + lr) * 2048;
    const __bf16* krow = K + (size_t)(q0 + lr) * 2048;
    v8f acc = {};
    for (int c0 = 0; c0 < 2048; c0 += 32) {
      BF16Frag a, b;
#pragma unroll
      for (int i = 0; i < 8; ++i) {
        a.h[i]     = qrow[c0 + hi * 8 + i];
        a.h[8 + i] = qrow[c0 + 16 + hi * 8 + i];
      }
#pragma unroll
      for (int i = 0; i < 16; ++i) b.h[i] = krow[c0 + hi * 8 + i + (i & 8)];
      acc = wmma_bf16(a.v, b.v, acc);
    }
#pragma unroll
    for (int v = 0; v < 8; ++v) {
      const int p = p0 + v + 8 * hi, q = q0 + lr;
      const float e = __expf(acc[v] * inv_scale - key_f32(rowmaxK[p]));
      attn[p * 256 + q] = (__bf16)e;
    }
  }
  __syncthreads();

  // Deterministic row sums: one thread per row.
  {
    float s = 0.0f;
    const __bf16* arow = attn + tid * 256;
    for (int q = 0; q < 256; ++q) s += (float)arow[q];
    rowsum[tid] = s;
  }
  __syncthreads();

  // Pass 3: O = attn . V. V staged in 256-column slabs via the TDM.
  for (int cb = 0; cb < 2048; cb += 256) {
    if (wave == 0) {
      // 2D tile: 256 rows x 256 bf16 cols from V[:, cb..cb+256)
      tdm_load_tile_bf16(V + cb, VSTAGE_OFF, 256, 256, 2048, 2048);
      __builtin_amdgcn_s_wait_tensorcnt(0);
    }
    __syncthreads();  // TDM data visible to all waves

    for (int t = wave; t < 256; t += 8) {   // 16 p-tiles x 16 c-tiles
      const int p0 = (t >> 4) * 16, cl = (t & 15) * 16;
      const __bf16* arow = attn + (p0 + lr) * 256;
      v8f acc = {};
      for (int k0 = 0; k0 < 256; k0 += 32) {
        BF16Frag a, b;
#pragma unroll
        for (int i = 0; i < 8; ++i) {
          a.h[i]     = arow[k0 + hi * 8 + i];
          a.h[8 + i] = arow[k0 + 16 + hi * 8 + i];
        }
#pragma unroll
        for (int i = 0; i < 16; ++i)        // B[k=q][n=c] from LDS stage
          b.h[i] = vst[(k0 + hi * 8 + i + (i & 8)) * 256 + cl + lr];
        acc = wmma_bf16(a.v, b.v, acc);
      }
      const int c = cb + cl + lr, xy = c >> 5, dk = c & 31, d = dk * 8 + head;
#pragma unroll
      for (int v = 0; v < 8; ++v) {
        const int p = p0 + v + 8 * hi;
        const float o = acc[v] / rowsum[p];
        const size_t oidx = (((size_t)bb * 256 + p) * 64 + xy) * 256 + d;
        Am[oidx] = (__bf16)o;
      }
    }
    __syncthreads();  // protect stage before next slab overwrite
  }
}

// ---------------------------------------------------------------------------
// Kernel 3: output projection. Am:[M,256] bf16, Wo bf16 from ws, out f32.
// ---------------------------------------------------------------------------
__global__ __launch_bounds__(256) void k_out_proj(
    const __bf16* __restrict__ Am,
    const __bf16* __restrict__ WoB, const float* __restrict__ bo,
    float* __restrict__ out)
{
  const int wave = threadIdx.x >> 5, lane = threadIdx.x & 31;
  const int lr = lane & 15, hi = lane >> 4;
  const int m0 = blockIdx.x * 128 + wave * 16;

  BF16Frag A[8];
  const __bf16* arow = Am + (size_t)(m0 + lr) * 256;
  __builtin_prefetch(arow, 0, 1);
#pragma unroll
  for (int kk = 0; kk < 8; ++kk) {
    const int kb = kk * 32 + hi * 8;
#pragma unroll
    for (int i = 0; i < 8; ++i) {
      A[kk].h[i]     = arow[kb + i];
      A[kk].h[8 + i] = arow[kb + 16 + i];
    }
  }
  for (int nt = 0; nt < 16; ++nt) {
    const int j = nt * 16 + lr;
    const __bf16* wrow = WoB + (size_t)j * 256 + hi * 8;
    v8f acc = {};
#pragma unroll
    for (int kk = 0; kk < 8; ++kk) {
      BF16Frag B;
#pragma unroll
      for (int i = 0; i < 16; ++i) B.h[i] = wrow[kk * 32 + i + (i & 8)];
      acc = wmma_bf16(A[kk].v, B.v, acc);
    }
    const float bias = bo[j];
#pragma unroll
    for (int v = 0; v < 8; ++v) {
      const size_t m = (size_t)(m0 + v + 8 * hi);
      out[m * 256 + j] = acc[v] + bias;
    }
  }
}

// ---------------------------------------------------------------------------
extern "C" void kernel_launch(void* const* d_in, const int* in_sizes, int n_in,
                              void* d_out, int out_size, void* d_ws, size_t ws_size,
                              hipStream_t stream) {
  (void)n_in; (void)out_size;
  const float* x  = (const float*)d_in[0];
  const float* Wq = (const float*)d_in[1];
  const float* bq = (const float*)d_in[2];
  const float* Wk = (const float*)d_in[3];
  const float* bk = (const float*)d_in[4];
  const float* Wv = (const float*)d_in[5];
  const float* bv = (const float*)d_in[6];
  const float* Wo = (const float*)d_in[7];
  const float* bo = (const float*)d_in[8];
  float* out = (float*)d_out;

  const size_t Mrows = (size_t)in_sizes[0] / 256;  // 131072
  const size_t elems = Mrows * 256;                // 33,554,432 per tensor

  // ws layout (bf16 elements): [Wq|Wk|Wv|Wo bf16: 4*65536][Qh][Kh][Vh][Am]
  __bf16* Wb = (__bf16*)d_ws;
  __bf16* WqB = Wb, *WkB = Wb + 65536, *WvB = Wb + 131072, *WoB = Wb + 196608;
  __bf16* Qh = Wb + 262144;
  __bf16* Kh = Qh + elems;
  __bf16* Vh = Kh + elems;
  // Dedicated region for merged attention output if workspace allows;
  // otherwise deterministically fall back to reusing the Q region.
  __bf16* Am = (ws_size >= (262144 + 4 * elems) * sizeof(__bf16))
                   ? (Vh + elems) : Qh;

  dim3 blk(256);
  k_cvt_weights<<<dim3(256), blk, 0, stream>>>(Wq, Wk, Wv, Wo, Wb);
  k_qkv_proj<<<dim3((unsigned)(Mrows / 128)), blk, 0, stream>>>(
      x, WqB, bq, WkB, bk, WvB, bv, Qh, Kh, Vh);
  k_attention<<<dim3(64), blk, LDS_BYTES, stream>>>(Qh, Kh, Vh, Am);
  k_out_proj<<<dim3((unsigned)(Mrows / 128)), blk, 0, stream>>>(Am, WoB, bo, out);
}
